// ConvLayer_65051574665680
// MI455X (gfx1250) — compile-verified, matrix operand order checked by reading, not run on hardware
//
#include <hip/hip_runtime.h>
#include <stdint.h>

typedef __attribute__((ext_vector_type(16))) _Float16 v16h;
typedef __attribute__((ext_vector_type(8)))  _Float16 v8h;
typedef __attribute__((ext_vector_type(8)))  float    v8f;
typedef __attribute__((ext_vector_type(4)))  float    vf4;
typedef __attribute__((ext_vector_type(4)))  int      i32x4;

#define NNODES 8192
#define NEDGES 65536
#define FIN    256
#define FOUT   64
#define EDIM   32

// f16 weight pool layout (element offsets)
#define WP_OFF 0                     // [64,256]  = 16384
#define WS_OFF 16384                 // [64,64]   =  4096
#define WN_OFF 20480                 // [64,64]   =  4096
#define WE_OFF 24576                 // [4096,32] = 131072
#define WH_TOTAL 155648

// ---------------------------------------------------------------------------
// gfx1250 async global->LDS copy (ASYNCcnt path), with portable fallback
// ---------------------------------------------------------------------------
#if defined(__has_builtin)
# if __has_builtin(__builtin_amdgcn_global_load_async_to_lds_b128) && \
     __has_builtin(__builtin_amdgcn_s_wait_asynccnt)
#  define USE_ASYNC_LDS 1
# endif
#endif
#ifndef USE_ASYNC_LDS
# define USE_ASYNC_LDS 0
#endif

#define AS1 __attribute__((address_space(1)))
#define AS3 __attribute__((address_space(3)))

__device__ __forceinline__ void lds_copy_b128(_Float16* dst,
                                              const _Float16* src) {
#if USE_ASYNC_LDS
  // Builtin prototype (from compiler diagnostic): param0 = global int4*,
  // param1 = LDS int4*, then imm offset, imm cpol. AS1 via 64-bit inttoptr;
  // AS3 via low-32 truncation (generic LDS address low bits are the
  // wave-relative LDS offset per the flat-aperture rules).
  __builtin_amdgcn_global_load_async_to_lds_b128(
      (AS1 i32x4*)(uintptr_t)(const void*)src,
      (AS3 i32x4*)(unsigned)(uintptr_t)(const void*)dst, 0, 0);
#else
  *(v8h*)dst = *(const v8h*)src;
#endif
}

// Copy `halfs` f16 elements (multiple of 2048) global -> LDS with 256 threads.
// Caller must __syncthreads() afterwards.
__device__ __forceinline__ void stage_lds(_Float16* dst, const _Float16* src,
                                          int halfs, int tid) {
  for (int i = tid * 8; i < halfs; i += 256 * 8) lds_copy_b128(dst + i, src + i);
#if USE_ASYNC_LDS
  __builtin_amdgcn_s_wait_asynccnt(0);
#endif
}

// ---------------------------------------------------------------------------
// WMMA + operand helpers
// ---------------------------------------------------------------------------
__device__ __forceinline__ v8f wmma_f16(v16h a, v16h b, v8f c) {
  return __builtin_amdgcn_wmma_f32_16x16x32_f16(false, a, false, b, (short)0, c,
                                                false, false);
}

// halfs [0..7] = K {0..7}, [8..15] = K {16..23} relative to p (sel*8 applied)
__device__ __forceinline__ v16h cvt_strips(const float* __restrict__ p) {
  vf4 a0 = *(const vf4*)(p);
  vf4 a1 = *(const vf4*)(p + 4);
  vf4 a2 = *(const vf4*)(p + 16);
  vf4 a3 = *(const vf4*)(p + 20);
  v16h r;
#pragma unroll
  for (int i = 0; i < 4; ++i) {
    r[i]      = (_Float16)a0[i];
    r[4 + i]  = (_Float16)a1[i];
    r[8 + i]  = (_Float16)a2[i];
    r[12 + i] = (_Float16)a3[i];
  }
  return r;
}

__device__ __forceinline__ v16h cvt_strips_scaled(const float* __restrict__ p,
                                                  float sc) {
  vf4 a0 = *(const vf4*)(p);
  vf4 a1 = *(const vf4*)(p + 4);
  vf4 a2 = *(const vf4*)(p + 16);
  vf4 a3 = *(const vf4*)(p + 20);
  v16h r;
#pragma unroll
  for (int i = 0; i < 4; ++i) {
    r[i]      = (_Float16)(a0[i] * sc);
    r[4 + i]  = (_Float16)(a1[i] * sc);
    r[8 + i]  = (_Float16)(a2[i] * sc);
    r[12 + i] = (_Float16)(a3[i] * sc);
  }
  return r;
}

__device__ __forceinline__ v16h lds_strips(const _Float16* bp) {
  v8h lo = *(const v8h*)(bp);
  v8h hi = *(const v8h*)(bp + 16);
  return __builtin_shufflevector(lo, hi, 0, 1, 2, 3, 4, 5, 6, 7, 8, 9, 10, 11,
                                 12, 13, 14, 15);
}

// ---------------------------------------------------------------------------
// kernel 0a: zero scratch (agg + deg)
// ---------------------------------------------------------------------------
__global__ void zero_kernel(float* __restrict__ p, int n) {
  int i = blockIdx.x * blockDim.x + threadIdx.x;
  if (i < n) p[i] = 0.0f;
}

// ---------------------------------------------------------------------------
// kernel 0b: convert all weights to f16 once (enables async LDS staging)
// ---------------------------------------------------------------------------
__global__ void cvt_weights_kernel(const float* __restrict__ wp,
                                   const float* __restrict__ ws,
                                   const float* __restrict__ wn,
                                   const float* __restrict__ we,
                                   _Float16* __restrict__ out) {
  int i = blockIdx.x * blockDim.x + threadIdx.x;
  if (i < WS_OFF)            out[i] = (_Float16)wp[i - WP_OFF];
  else if (i < WN_OFF)       out[i] = (_Float16)ws[i - WS_OFF];
  else if (i < WE_OFF)       out[i] = (_Float16)wn[i - WN_OFF];
  else if (i < WH_TOTAL)     out[i] = (_Float16)we[i - WE_OFF];
}

// ---------------------------------------------------------------------------
// kernel 1: hn = relu(h_neigh @ Wp^T), hs = relu(h_self @ Wp^T)
// grid (64, 2), block 256; wave = 16 rows x 64 cols, K = 256
// ---------------------------------------------------------------------------
__global__ __launch_bounds__(256) void preagg_kernel(
    const float* __restrict__ Hn, const float* __restrict__ Hs,
    const _Float16* __restrict__ wp_h,  // [64,256] f16
    float* __restrict__ hn_out, float* __restrict__ hs_out) {
  __shared__ _Float16 sW[FOUT * FIN];  // 32 KB
  const int tid = threadIdx.x;
  stage_lds(sW, wp_h, FOUT * FIN, tid);
  __syncthreads();

  const float* A = (blockIdx.y == 0) ? Hn : Hs;
  float* Out     = (blockIdx.y == 0) ? hn_out : hs_out;

  const int wave = tid >> 5, lane = tid & 31;
  const int r = lane & 15, sel = lane >> 4;
  const int mBase = blockIdx.x * 128 + wave * 16;
  const int row = mBase + r;

  v8f acc[4] = {v8f{}, v8f{}, v8f{}, v8f{}};
  for (int k0 = 0; k0 < FIN; k0 += 32) {
    v16h a = cvt_strips(A + (size_t)row * FIN + k0 + sel * 8);
#pragma unroll
    for (int nt = 0; nt < 4; ++nt) {
      v16h b = lds_strips(&sW[(size_t)(nt * 16 + r) * FIN + k0 + sel * 8]);
      acc[nt] = wmma_f16(a, b, acc[nt]);
    }
  }
#pragma unroll
  for (int nt = 0; nt < 4; ++nt) {
#pragma unroll
    for (int v = 0; v < 8; ++v) {
      int m = mBase + sel * 8 + v;
      Out[(size_t)m * FOUT + nt * 16 + r] = fmaxf(acc[nt][v], 0.0f);
    }
  }
}

// ---------------------------------------------------------------------------
// kernel 2: edge GEMM [E,32]x[32,4096] + relu + fold i-sum -> s[e,64];
// msg = hn[src]*s -> atomic segment-sum; degree count.
// block 256 (8 waves); each wave owns 32 edges (2 A-tiles) so every B tile
// staged in LDS feeds 2 WMMAs (halves ds bandwidth per FLOP). 256 blocks.
// W_edge f16 streamed via async copies in 4 chunks of 64 KB.
// ---------------------------------------------------------------------------
__global__ __launch_bounds__(256) void edge_kernel(
    const float* __restrict__ EF,        // [E,32]
    const _Float16* __restrict__ we_h,   // [4096,32] f16
    const float* __restrict__ be,        // [4096]
    const int* __restrict__ src, const int* __restrict__ dst,
    const float* __restrict__ hn,        // [N,64]
    float* __restrict__ agg,             // [N,64]
    float* __restrict__ deg) {           // [N]
  __shared__ _Float16 sB[1024 * EDIM];   // 64 KB chunk, [c][k] f16

  const int tid = threadIdx.x, wave = tid >> 5, lane = tid & 31;
  const int r = lane & 15, sel = lane >> 4;
  const int eBase = blockIdx.x * 256 + wave * 32;

  // Two 16-edge A tiles, loaded once (K = 32 = one WMMA step)
  v16h a0 = cvt_strips(EF + (size_t)(eBase + r) * EDIM + sel * 8);
  v16h a1 = cvt_strips(EF + (size_t)(eBase + 16 + r) * EDIM + sel * 8);

  v8f s0[4] = {v8f{}, v8f{}, v8f{}, v8f{}};
  v8f s1[4] = {v8f{}, v8f{}, v8f{}, v8f{}};

  for (int chunk = 0; chunk < 4; ++chunk) {
    __syncthreads();  // previous chunk fully consumed
    stage_lds(sB, we_h + (size_t)chunk * 1024 * EDIM, 1024 * EDIM, tid);
    __syncthreads();

    for (int t = 0; t < 16; ++t) {
#pragma unroll
      for (int q = 0; q < 4; ++q) {            // nTile % 4 == q (compile-time)
        const int nTile = chunk * 64 + t * 4 + q;
        const int c = nTile * 16 + r;          // global output column
        const float bb = be[c];
        v8f cinit = {bb, bb, bb, bb, bb, bb, bb, bb};
        v16h b =
            lds_strips(&sB[(size_t)((t * 4 + q) * 16 + r) * EDIM + sel * 8]);
        v8f d0 = wmma_f16(a0, b, cinit);
        v8f d1 = wmma_f16(a1, b, cinit);
#pragma unroll
        for (int v = 0; v < 8; ++v) {
          s0[q][v] += fmaxf(d0[v], 0.0f);      // relu + fold i-sum
          s1[q][v] += fmaxf(d1[v], 0.0f);
        }
      }
    }
  }

  // msg[e,j] = hn[src[e],j] * s[e,j]; scatter-add into agg[dst[e],j]
#pragma unroll
  for (int q = 0; q < 4; ++q) {
    const int j = q * 16 + r;
#pragma unroll
    for (int v = 0; v < 8; ++v) {
      {
        const int e = eBase + sel * 8 + v;
        const int se = src[e], de = dst[e];
        atomicAdd(&agg[(size_t)de * FOUT + j],
                  hn[(size_t)se * FOUT + j] * s0[q][v]);
      }
      {
        const int e = eBase + 16 + sel * 8 + v;
        const int se = src[e], de = dst[e];
        atomicAdd(&agg[(size_t)de * FOUT + j],
                  hn[(size_t)se * FOUT + j] * s1[q][v]);
      }
    }
  }
  if (r == 0) {  // lanes 0 & 16 cover this wave's 32 edges
#pragma unroll
    for (int v = 0; v < 8; ++v) {
      atomicAdd(&deg[dst[eBase + sel * 8 + v]], 1.0f);
      atomicAdd(&deg[dst[eBase + 16 + sel * 8 + v]], 1.0f);
    }
  }
}

// ---------------------------------------------------------------------------
// kernel 3: neigh = agg/max(deg,1); z = relu(relu(hs@Ws^T)+relu(neigh@Wn^T))
// grid 64, block 256; per wave 16 rows x 64 cols, K = 64
// ---------------------------------------------------------------------------
__global__ __launch_bounds__(256) void out_kernel(
    const float* __restrict__ hs,        // [N,64]
    const float* __restrict__ agg,       // [N,64]
    const float* __restrict__ deg,       // [N]
    const _Float16* __restrict__ ws_h,   // [64,64] f16
    const _Float16* __restrict__ wn_h,   // [64,64] f16
    float* __restrict__ z) {             // [N,64]
  __shared__ _Float16 sWs[FOUT * FOUT];  // 8 KB
  __shared__ _Float16 sWn[FOUT * FOUT];  // 8 KB
  const int tid = threadIdx.x;
  stage_lds(sWs, ws_h, FOUT * FOUT, tid);
  stage_lds(sWn, wn_h, FOUT * FOUT, tid);
  __syncthreads();

  const int wave = tid >> 5, lane = tid & 31;
  const int r = lane & 15, sel = lane >> 4;
  const int mBase = blockIdx.x * 128 + wave * 16;
  const int row = mBase + r;
  const float invd = 1.0f / fmaxf(deg[row], 1.0f);

  v8f accS[4] = {v8f{}, v8f{}, v8f{}, v8f{}};
  v8f accN[4] = {v8f{}, v8f{}, v8f{}, v8f{}};
  for (int k0 = 0; k0 < FOUT; k0 += 32) {
    v16h aS = cvt_strips(hs + (size_t)row * FOUT + k0 + sel * 8);
    v16h aN = cvt_strips_scaled(agg + (size_t)row * FOUT + k0 + sel * 8, invd);
#pragma unroll
    for (int nt = 0; nt < 4; ++nt) {
      v16h bS = lds_strips(&sWs[(size_t)(nt * 16 + r) * FOUT + k0 + sel * 8]);
      v16h bN = lds_strips(&sWn[(size_t)(nt * 16 + r) * FOUT + k0 + sel * 8]);
      accS[nt] = wmma_f16(aS, bS, accS[nt]);
      accN[nt] = wmma_f16(aN, bN, accN[nt]);
    }
  }
#pragma unroll
  for (int nt = 0; nt < 4; ++nt) {
#pragma unroll
    for (int v = 0; v < 8; ++v) {
      const int m = mBase + sel * 8 + v;
      const float val = fmaxf(accS[nt][v], 0.0f) + fmaxf(accN[nt][v], 0.0f);
      z[(size_t)m * FOUT + nt * 16 + r] = fmaxf(val, 0.0f);
    }
  }
}

// ---------------------------------------------------------------------------
// launch
// ---------------------------------------------------------------------------
extern "C" void kernel_launch(void* const* d_in, const int* in_sizes, int n_in,
                              void* d_out, int out_size, void* d_ws,
                              size_t ws_size, hipStream_t stream) {
  const float* h_neigh  = (const float*)d_in[0];
  const float* h_self   = (const float*)d_in[1];
  const float* edge_f   = (const float*)d_in[2];
  const float* W_preagg = (const float*)d_in[3];
  const float* W_self   = (const float*)d_in[4];
  const float* W_neigh  = (const float*)d_in[5];
  const float* W_edge   = (const float*)d_in[6];
  const float* b_edge   = (const float*)d_in[7];
  const int*   src      = (const int*)d_in[8];
  const int*   dst      = (const int*)d_in[9];
  float* out = (float*)d_out;

  float* hn  = (float*)d_ws;                    // [8192,64]
  float* hs  = hn + (size_t)NNODES * FOUT;      // [8192,64]
  float* agg = hs + (size_t)NNODES * FOUT;      // [8192,64]
  float* deg = agg + (size_t)NNODES * FOUT;     // [8192]
  _Float16* wh = (_Float16*)(deg + NNODES);     // f16 weight pool (16B aligned)

  const int nz = NNODES * FOUT + NNODES;        // zero agg + deg every call
  zero_kernel<<<(nz + 255) / 256, 256, 0, stream>>>(agg, nz);

  cvt_weights_kernel<<<(WH_TOTAL + 255) / 256, 256, 0, stream>>>(
      W_preagg, W_self, W_neigh, W_edge, wh);

  preagg_kernel<<<dim3(NNODES / 128, 2), 256, 0, stream>>>(
      h_neigh, h_self, wh + WP_OFF, hn, hs);

  edge_kernel<<<NEDGES / 256, 256, 0, stream>>>(edge_f, wh + WE_OFF, b_edge,
                                                src, dst, hn, agg, deg);

  out_kernel<<<NNODES / 128, 256, 0, stream>>>(hs, agg, deg, wh + WS_OFF,
                                               wh + WN_OFF, out);
}
